// MultiHeadAttention_7060926234889
// MI455X (gfx1250) — matrix operand order repaired and checked by reading, not compiled, not run on hardware
//
#include <hip/hip_runtime.h>
#include <hip/hip_bf16.h>

typedef __attribute__((ext_vector_type(16))) _Float16 v16h;
typedef __attribute__((ext_vector_type(8)))  _Float16 v8h;
typedef __attribute__((ext_vector_type(8)))  float    v8f;

#define BB 4
#define SS 2048
#define DD 1024
#define HH 16
#define HD 64
#define MM (BB * SS)   // 8192 total rows (B*S)

// ---------------------------------------------------------------------------
// Load 16 f16 values as a WMMA operand half-pair: 8 halves from p0, 8 from p1.
// 16B-aligned in all our layouts -> two global_load_b128 / ds_load_b128.
// ---------------------------------------------------------------------------
__device__ __forceinline__ v16h ld16(const _Float16* p0, const _Float16* p1) {
  v8h lo = *(const v8h*)p0;
  v8h hi = *(const v8h*)p1;
  v16h r;
#pragma unroll
  for (int i = 0; i < 8; ++i) { r[i] = lo[i]; r[8 + i] = hi[i]; }
  return r;
}

__device__ __forceinline__ v8f wmma_f16(v16h a, v16h b, v8f c) {
  // D = A(16x32 f16) * B(32x16 f16) + C(16x16 f32)
  return __builtin_amdgcn_wmma_f32_16x16x32_f16(false, a, false, b, (short)0, c,
                                                false, false);
}

// ---------------------------------------------------------------------------
// f32 -> f16 conversion (grid-stride)
// ---------------------------------------------------------------------------
__global__ void cvt_f32_f16(const float* __restrict__ s, _Float16* __restrict__ d, int n) {
  int i = blockIdx.x * blockDim.x + threadIdx.x;
  int stride = gridDim.x * blockDim.x;
  for (; i < n; i += stride) d[i] = (_Float16)s[i];
}

// ---------------------------------------------------------------------------
// Register-blocked GEMM: out = A[M,K](f16) @ W[N,K]^T(f16) + bias[N](f32)
// One wave -> one 64x64 output block (4x4 grid of 16x16 WMMA tiles).
// Per K-step (32): 4 A operands + 4 B operands (16 b128 loads) -> 16 v_wmma.
// mode 0: f32 out, row-major [M,N]          (final projection -> d_out)
// mode 1: f16 out, [B,H,S,hd] head-split    (Q and K)
// mode 2: f16 out, [B,H,hd,S] transposed    (V, so PV B-operand is contiguous)
// ---------------------------------------------------------------------------
__global__ __launch_bounds__(32) void gemm64(const _Float16* __restrict__ A,
                                             const _Float16* __restrict__ W,
                                             const float* __restrict__ bias,
                                             float* __restrict__ outF,
                                             _Float16* __restrict__ outH,
                                             int Kdim, int Ndim, int mode) {
  const int lane = threadIdx.x & 31;
  const int l16  = lane & 15;
  const int hi   = lane >> 4;            // 0 or 1
  const int nt   = blockIdx.x;           // N / 64
  const int mt   = blockIdx.y;           // M / 64

  const _Float16* Ap[4];
  const _Float16* Wp[4];
#pragma unroll
  for (int i = 0; i < 4; ++i) Ap[i] = A + (size_t)(mt * 64 + i * 16 + l16) * Kdim;
#pragma unroll
  for (int j = 0; j < 4; ++j) Wp[j] = W + (size_t)(nt * 64 + j * 16 + l16) * Kdim;

  v8f acc[4][4] = {};
  for (int k0 = 0; k0 < Kdim; k0 += 32) {
    v16h a[4], b[4];
    // A: slot i -> K = k0 + (i/8)*16 + hi*8 + i%8
#pragma unroll
    for (int i = 0; i < 4; ++i)
      a[i] = ld16(Ap[i] + k0 + hi * 8, Ap[i] + k0 + 16 + hi * 8);
    // B: slot i -> K = k0 + (i/8)*16 + hi*8 + i%8
#pragma unroll
    for (int j = 0; j < 4; ++j)
      b[j] = ld16(Wp[j] + k0 + hi * 8, Wp[j] + k0 + 16 + hi * 8);
#pragma unroll
    for (int i = 0; i < 4; ++i)
#pragma unroll
      for (int j = 0; j < 4; ++j)
        acc[i][j] = wmma_f16(a[i], b[j], acc[i][j]);
  }

#pragma unroll
  for (int j = 0; j < 4; ++j) {
    const int ncol = nt * 64 + j * 16 + l16;      // column held by this lane (C layout)
    const float bval = bias[ncol];
#pragma unroll
    for (int i = 0; i < 4; ++i) {
#pragma unroll
      for (int r = 0; r < 8; ++r) {
        // C layout: VGPR r holds row (r + 8*hi) of the 16x16 tile
        const int orow = mt * 64 + i * 16 + r + hi * 8;
        const float v = acc[i][j][r] + bval;
        if (mode == 0) {
          outF[(size_t)orow * Ndim + ncol] = v;
        } else {
          const int bidx = orow / SS, srow = orow % SS;
          const int h = ncol / HD, dh = ncol % HD;
          if (mode == 1)
            outH[(((size_t)bidx * HH + h) * SS + srow) * HD + dh] = (_Float16)v;
          else
            outH[(((size_t)bidx * HH + h) * HD + dh) * SS + srow] = (_Float16)v;
        }
      }
    }
  }
}

// ---------------------------------------------------------------------------
// Flash attention: one wave handles 32 query rows (two 16-row tiles) of one
// (b,h), so each loaded K/V tile feeds two q-tiles (16 wmma per 16 loads).
// Q,K [bh][S][64] f16, Vt [bh][64][S] f16 -> Ctx [B*S][D] f16
// ---------------------------------------------------------------------------
__global__ __launch_bounds__(32) void attn32(const _Float16* __restrict__ Q,
                                             const _Float16* __restrict__ K,
                                             const _Float16* __restrict__ Vt,
                                             _Float16* __restrict__ Ctx) {
  __shared__ _Float16 pbuf[32 * 32];     // P pivot: C-layout f32 -> A-layout f16

  const int lane = threadIdx.x & 31;
  const int l16  = lane & 15;
  const int hi   = lane >> 4;
  const int qt   = blockIdx.x & (SS / 32 - 1);   // 64 q-blocks per head
  const int bh   = blockIdx.x >> 6;

  const _Float16* Kp = K + (size_t)bh * SS * HD;
  const _Float16* Vp = Vt + (size_t)bh * HD * SS;

  // Q block as 2 tiles x 2 K-steps of A operands, loaded once
  v16h qa[2][2];
#pragma unroll
  for (int t = 0; t < 2; ++t) {
    const _Float16* Qp = Q + (size_t)bh * SS * HD +
                         (size_t)(qt * 32 + t * 16 + l16) * HD;
    qa[t][0] = ld16(Qp + hi * 8,      Qp + 16 + hi * 8);
    qa[t][1] = ld16(Qp + 32 + hi * 8, Qp + 48 + hi * 8);
  }

  v8f c[2][4] = {};                      // ctx accumulators: [q-tile][hd chunk]
  float mrow[2][8], lrow[2][8];
#pragma unroll
  for (int t = 0; t < 2; ++t)
#pragma unroll
    for (int r = 0; r < 8; ++r) { mrow[t][r] = -3.0e38f; lrow[t][r] = 0.f; }

  for (int kb = 0; kb < SS; kb += 32) {
    // ---- K tile B-operands (shared by both q-tiles) ----
    v16h kbop[2][2];
#pragma unroll
    for (int nc = 0; nc < 2; ++nc) {
      const _Float16* kr = Kp + (size_t)(kb + nc * 16 + l16) * HD;
      kbop[nc][0] = ld16(kr + hi * 8,      kr + 16 + hi * 8);
      kbop[nc][1] = ld16(kr + 32 + hi * 8, kr + 48 + hi * 8);
    }
    // ---- scores: 2 q-tiles x 2 key chunks ----
    v8f s[2][2] = {};
#pragma unroll
    for (int t = 0; t < 2; ++t)
#pragma unroll
      for (int nc = 0; nc < 2; ++nc) {
        s[t][nc] = wmma_f16(qa[t][0], kbop[nc][0], s[t][nc]);
        s[t][nc] = wmma_f16(qa[t][1], kbop[nc][1], s[t][nc]);
      }

    // ---- online softmax; row M = r + 8*hi lives across a 16-lane group ----
#pragma unroll
    for (int t = 0; t < 2; ++t) {
#pragma unroll
      for (int r = 0; r < 8; ++r) {
        float v0 = s[t][0][r] * 0.125f;           // 1/sqrt(64)
        float v1 = s[t][1][r] * 0.125f;
        float mx = fmaxf(v0, v1);
#pragma unroll
        for (int m = 1; m <= 8; m <<= 1) mx = fmaxf(mx, __shfl_xor(mx, m, 32));
        const float mnew = fmaxf(mrow[t][r], mx);
        const float resc = __expf(mrow[t][r] - mnew);
        const float p0 = __expf(v0 - mnew);
        const float p1 = __expf(v1 - mnew);
        float rs = p0 + p1;
#pragma unroll
        for (int m = 1; m <= 8; m <<= 1) rs += __shfl_xor(rs, m, 32);
        lrow[t][r] = lrow[t][r] * resc + rs;
        mrow[t][r] = mnew;
        c[t][0][r] *= resc; c[t][1][r] *= resc;
        c[t][2][r] *= resc; c[t][3][r] *= resc;
        pbuf[(t * 16 + r + hi * 8) * 32 + l16]      = (_Float16)p0;
        pbuf[(t * 16 + r + hi * 8) * 32 + 16 + l16] = (_Float16)p1;
      }
    }
    __syncthreads();
    // re-read P as A operands (16x32): slot i -> K = (i/8)*16 + hi*8 + i%8
    v16h pa[2];
#pragma unroll
    for (int t = 0; t < 2; ++t)
      pa[t] = ld16(&pbuf[(t * 16 + l16) * 32 + hi * 8],
                   &pbuf[(t * 16 + l16) * 32 + 16 + hi * 8]);
    __syncthreads();

    // ---- ctx += P(16x32) @ V(32x64); each V operand feeds both q-tiles ----
#pragma unroll
    for (int nc = 0; nc < 4; ++nc) {
      const _Float16* vp = Vp + (size_t)(nc * 16 + l16) * SS + kb + hi * 8;
      const v16h vb = ld16(vp, vp + 16);
      c[0][nc] = wmma_f16(pa[0], vb, c[0][nc]);
      c[1][nc] = wmma_f16(pa[1], vb, c[1][nc]);
    }
  }

  // ---- epilogue: ctx / l, merge heads into [B*S][D] f16 ----
  const int b = bh / HH, h = bh % HH;
#pragma unroll
  for (int t = 0; t < 2; ++t) {
#pragma unroll
    for (int r = 0; r < 8; ++r) {
      const float inv = 1.0f / lrow[t][r];
      const size_t orow = (size_t)b * SS + (size_t)qt * 32 + t * 16 + r + hi * 8;
      _Float16* op = Ctx + orow * DD + h * HD;
#pragma unroll
      for (int nc = 0; nc < 4; ++nc)
        op[nc * 16 + l16] = (_Float16)(c[t][nc][r] * inv);
    }
  }
}

// ---------------------------------------------------------------------------
extern "C" void kernel_launch(void* const* d_in, const int* in_sizes, int n_in,
                              void* d_out, int out_size, void* d_ws, size_t ws_size,
                              hipStream_t stream) {
  (void)in_sizes; (void)n_in; (void)out_size; (void)ws_size;
  const float* x  = (const float*)d_in[0];
  const float* Wq = (const float*)d_in[1];
  const float* bq = (const float*)d_in[2];
  const float* Wk = (const float*)d_in[3];
  const float* bk = (const float*)d_in[4];
  const float* Wv = (const float*)d_in[5];
  const float* bv = (const float*)d_in[6];
  const float* Wo = (const float*)d_in[7];
  const float* bo = (const float*)d_in[8];
  float* out = (float*)d_out;

  char* ws = (char*)d_ws;
  size_t off = 0;
  auto take = [&](size_t bytes) -> char* {
    char* p = ws + off;
    off += (bytes + 255) & ~(size_t)255;
    return p;
  };
  _Float16* xh  = (_Float16*)take((size_t)MM * DD * 2);   // x in f16
  _Float16* wqh = (_Float16*)take((size_t)DD * DD * 2);
  _Float16* wkh = (_Float16*)take((size_t)DD * DD * 2);
  _Float16* wvh = (_Float16*)take((size_t)DD * DD * 2);
  _Float16* woh = (_Float16*)take((size_t)DD * DD * 2);
  _Float16* qh  = (_Float16*)take((size_t)MM * DD * 2);   // [B,H,S,hd]
  _Float16* kh  = (_Float16*)take((size_t)MM * DD * 2);   // [B,H,S,hd]
  _Float16* vth = (_Float16*)take((size_t)MM * DD * 2);   // [B,H,hd,S]
  _Float16* ch  = (_Float16*)take((size_t)MM * DD * 2);   // ctx, [B*S][D]

  // 1) precision conversion
  cvt_f32_f16<<<dim3(2048), dim3(256), 0, stream>>>(x,  xh,  MM * DD);
  cvt_f32_f16<<<dim3(512),  dim3(256), 0, stream>>>(Wq, wqh, DD * DD);
  cvt_f32_f16<<<dim3(512),  dim3(256), 0, stream>>>(Wk, wkh, DD * DD);
  cvt_f32_f16<<<dim3(512),  dim3(256), 0, stream>>>(Wv, wvh, DD * DD);
  cvt_f32_f16<<<dim3(512),  dim3(256), 0, stream>>>(Wo, woh, DD * DD);

  // 2) QKV projections (register-blocked WMMA GEMMs, 64x64 per wave)
  dim3 ggrid(DD / 64, MM / 64);   // 16 x 128 waves
  gemm64<<<ggrid, dim3(32), 0, stream>>>(xh, wqh, bq, nullptr, qh,  DD, DD, 1);
  gemm64<<<ggrid, dim3(32), 0, stream>>>(xh, wkh, bk, nullptr, kh,  DD, DD, 1);
  gemm64<<<ggrid, dim3(32), 0, stream>>>(xh, wvh, bv, nullptr, vth, DD, DD, 2);

  // 3) flash attention: one wave per 32 query rows per (b,h)
  attn32<<<dim3(BB * HH * (SS / 32)), dim3(32), 0, stream>>>(qh, kh, vth, ch);

  // 4) output projection -> f32 d_out
  gemm64<<<ggrid, dim3(32), 0, stream>>>(ch, woh, bo, out, nullptr, DD, DD, 0);
}
